// Attention_18399639896530
// MI455X (gfx1250) — matrix operand order repaired and checked by reading, not compiled, not run on hardware
//
#include <hip/hip_runtime.h>

typedef __attribute__((ext_vector_type(2)))  float    v2f;
typedef __attribute__((ext_vector_type(8)))  float    v8f;
typedef __attribute__((ext_vector_type(8)))  _Float16 v8h;
typedef __attribute__((ext_vector_type(16))) _Float16 v16h;

#define B_   2
#define L_   4096
#define C_   256
#define H_   4
#define D_   32
#define O3_  384
#define HID_ 128

// ---- wave32 cross-lane reductions over the 16-lane half-groups (n axis of a
// 16x16 C/D fragment).  xor masks 1,2,4,8 never cross bit4, so lanes 0-15 and
// 16-31 reduce independently (they hold rows r and r+8 respectively).
__device__ __forceinline__ float grp16_max(float v) {
  v = fmaxf(v, __shfl_xor(v, 1, 32));
  v = fmaxf(v, __shfl_xor(v, 2, 32));
  v = fmaxf(v, __shfl_xor(v, 4, 32));
  v = fmaxf(v, __shfl_xor(v, 8, 32));
  return v;
}
__device__ __forceinline__ float grp16_sum(float v) {
  v += __shfl_xor(v, 1, 32);
  v += __shfl_xor(v, 2, 32);
  v += __shfl_xor(v, 4, 32);
  v += __shfl_xor(v, 8, 32);
  return v;
}

// ============================================================================
// Kernel 1: QKV projection  G[b][o][l] = sum_c w_qkv[o][c] * x[b][c][l]
// One wave per 16x16 output tile, v_wmma_f32_16x16x4_f32 over K=256.
// waves = B_ * (384/16) * (4096/16) = 12288 -> 1536 blocks of 8 waves.
// ============================================================================
__global__ __launch_bounds__(256) void qkv_gemm_kernel(const float* __restrict__ x,
                                                       const float* __restrict__ w,
                                                       float* __restrict__ G) {
  const int wg   = blockIdx.x * 8 + (threadIdx.x >> 5);
  const int lane = threadIdx.x & 31;
  const int m    = lane & 15;
  const int hi   = lane >> 4;
  const int lt   = wg & 255;
  const int tmp  = wg >> 8;            // 0..47
  const int ot   = tmp % 24;
  const int b    = tmp / 24;

  // A[m][k] = w[ot*16+m][k]; lane needs K = k0 + hi*2 + {0,1} -> float2 load
  const float* arow = w + (ot * 16 + m) * C_ + hi * 2;
  // B[k][n] = x[b][k][lt*16+n], n == m for this lane; stride L_ between k
  const float* bcol = x + (size_t)b * C_ * L_ + (lt * 16 + m);

  v8f acc = {};
#pragma unroll 4
  for (int k0 = 0; k0 < C_; k0 += 4) {
    v2f a = *(const v2f*)(arow + k0);
    v2f bb;
    bb[0] = bcol[(size_t)(k0 + hi * 2) * L_];
    bb[1] = bcol[(size_t)(k0 + hi * 2 + 1) * L_];
    acc = __builtin_amdgcn_wmma_f32_16x16x4_f32(false, a, false, bb,
                                                (short)0, acc, false, false);
  }
  // D layout: VGPR r -> row r + 8*hi, col = m
  float* grow = G + ((size_t)b * O3_ + ot * 16) * L_ + lt * 16 + m;
#pragma unroll
  for (int r = 0; r < 8; ++r)
    grow[(size_t)(r + 8 * hi) * L_] = acc[r];
}

// ============================================================================
// Kernel 2: inverse L2 norms over l for the 256 q/k rows per batch.
// blockIdx.x = b*256 + o, 256 threads reduce 4096 elements.
// ============================================================================
__global__ __launch_bounds__(256) void norm_kernel(const float* __restrict__ G,
                                                   float* __restrict__ invn) {
  __shared__ float red[256];
  const int tid = threadIdx.x;
  const int b   = blockIdx.x >> 8;
  const int o   = blockIdx.x & 255;
  const float* row = G + ((size_t)b * O3_ + o) * L_;
  float s = 0.f;
  for (int j = tid; j < L_; j += 256) { float v = row[j]; s += v * v; }
  red[tid] = s;
  __syncthreads();
  for (int off = 128; off > 0; off >>= 1) {
    if (tid < off) red[tid] += red[tid + off];
    __syncthreads();
  }
  if (tid == 0) invn[blockIdx.x] = 1.0f / fmaxf(sqrtf(red[0]), 1e-12f);
}

// ============================================================================
// Kernel 3: pack into f16 fragment-friendly layouts:
//   qt[bh][l][32]  (normalized q, transposed to l-major)
//   kt[bh][l][32]  (normalized k, transposed to l-major)
//   vv[bh][32][l]  (v, natural d-major)
// 1M threads, each handles one element of each tensor.
// ============================================================================
__global__ __launch_bounds__(256) void pack_kernel(const float* __restrict__ G,
                                                   const float* __restrict__ invn,
                                                   _Float16* __restrict__ qt,
                                                   _Float16* __restrict__ kt,
                                                   _Float16* __restrict__ vv) {
  const int gid = blockIdx.x * 256 + threadIdx.x;   // < 2^20
  // q/k mapping: gid = bh<<17 | l<<5 | d
  {
    const int d  = gid & 31;
    const int l  = (gid >> 5) & 4095;
    const int bh = gid >> 17;
    const int b  = bh >> 2, h = bh & 3;
    const int oq = h * 32 + d;
    qt[gid] = (_Float16)(G[((size_t)b * O3_ + oq) * L_ + l] * invn[b * 256 + oq]);
    kt[gid] = (_Float16)(G[((size_t)b * O3_ + 128 + oq) * L_ + l] * invn[b * 256 + 128 + oq]);
  }
  // v mapping: gid = bh<<17 | d<<12 | l  (contiguous read & write)
  {
    const int l  = gid & 4095;
    const int d  = (gid >> 12) & 31;
    const int bh = gid >> 17;
    const int b  = bh >> 2, h = bh & 3;
    vv[gid] = (_Float16)(G[((size_t)b * O3_ + 256 + h * 32 + d) * L_ + l]);
  }
}

// ============================================================================
// Kernel 4: flash attention, one wave = 16 query rows, j in chunks of 32.
// S = Q^T K via 2x v_wmma_f32_16x16x32_f16 (K dim = d_head = 32 exactly),
// online softmax in base-2 (scale 10*log2e folded in), P staged through
// per-wave LDS to convert C-layout -> A-layout, O += P V^T via 2 more WMMAs.
// grid = B_*H_*32 blocks of 8 waves (128 query rows per block).
// ============================================================================
__global__ __launch_bounds__(256) void flash_kernel(const _Float16* __restrict__ qt,
                                                    const _Float16* __restrict__ kt,
                                                    const _Float16* __restrict__ vv,
                                                    float* __restrict__ AO) {
  __shared__ __align__(16) _Float16 pbuf[8][16][32];   // per-wave P tile
  const int tid  = threadIdx.x;
  const int wid  = tid >> 5;
  const int lane = tid & 31;
  const int m    = lane & 15;
  const int hi   = lane >> 4;
  const int bh   = blockIdx.x >> 5;
  const int iblk = blockIdx.x & 31;
  const int i0   = iblk * 128 + wid * 16;

  const float cS = 10.0f * 1.44269504088896340736f;    // SCALE * log2(e)

  // ---- Q A-fragment: A[m][k=d]; lane m, K = hi*16 + 0..15 contiguous
  v16h aq;
  {
    const _Float16* qrow = qt + ((size_t)bh * L_ + i0 + m) * 32;
    v8h q0 = *(const v8h*)(qrow + hi * 16);
    v8h q1 = *(const v8h*)(qrow + hi * 16 + 8);
#pragma unroll
    for (int t = 0; t < 8; ++t) { aq[t] = q0[t]; aq[8 + t] = q1[t]; }
  }

  v8f oa = {}, ob = {};
  float rmax[8], rsum[8];
#pragma unroll
  for (int r = 0; r < 8; ++r) { rmax[r] = -3.0e38f; rsum[r] = 0.f; }

  const _Float16* kbase = kt + (size_t)bh * L_ * 32;
  const _Float16* vbase = vv + (size_t)bh * 32 * L_;

  for (int j0 = 0; j0 < L_; j0 += 32) {
    // ---- K B-fragments: B[k=d][n=j]; lane n=m, K = hi*16 + 0..15 contiguous
    v16h bka, bkb;
    {
      const _Float16* ka = kbase + (size_t)(j0 + m) * 32 + hi * 16;
      v8h t0 = *(const v8h*)ka, t1 = *(const v8h*)(ka + 8);
      const _Float16* kb2 = kbase + (size_t)(j0 + 16 + m) * 32 + hi * 16;
      v8h u0 = *(const v8h*)kb2, u1 = *(const v8h*)(kb2 + 8);
#pragma unroll
      for (int t = 0; t < 8; ++t) {
        bka[t] = t0[t]; bka[8 + t] = t1[t];
        bkb[t] = u0[t]; bkb[8 + t] = u1[t];
      }
    }
    v8f sa = {}, sb = {};
    sa = __builtin_amdgcn_wmma_f32_16x16x32_f16(false, aq, false, bka,
                                                (short)0, sa, false, false);
    sb = __builtin_amdgcn_wmma_f32_16x16x32_f16(false, aq, false, bkb,
                                                (short)0, sb, false, false);

    // ---- online softmax; fragment row = r + 8*hi, col = m (+16 for sb)
#pragma unroll
    for (int r = 0; r < 8; ++r) {
      float va = sa[r] * cS, vb = sb[r] * cS;
      float tm = grp16_max(fmaxf(va, vb));
      float nm = fmaxf(rmax[r], tm);
      float al = __builtin_amdgcn_exp2f(rmax[r] - nm);
      float pa = __builtin_amdgcn_exp2f(va - nm);
      float pb = __builtin_amdgcn_exp2f(vb - nm);
      rsum[r] = rsum[r] * al + grp16_sum(pa + pb);
      oa[r] *= al;
      ob[r] *= al;
      rmax[r] = nm;
      pbuf[wid][r + 8 * hi][m]      = (_Float16)pa;
      pbuf[wid][r + 8 * hi][16 + m] = (_Float16)pb;
    }
    // same-wave LDS write -> read; LDS is in-order per wave, just drain DScnt
    asm volatile("s_wait_dscnt 0x0" ::: "memory");

    // ---- P A-fragment (16x32): lane m, K = hi*16 + 0..15 contiguous
    v16h ap;
    {
      const _Float16* pr = &pbuf[wid][m][0];
      v8h p0 = *(const v8h*)(pr + hi * 16);
      v8h p1 = *(const v8h*)(pr + hi * 16 + 8);
#pragma unroll
      for (int t = 0; t < 8; ++t) { ap[t] = p0[t]; ap[8 + t] = p1[t]; }
    }
    // ---- V B-fragments: B[k=j][n=d]; lane n=m (d) / m+16, K = hi*16 + 0..15
    v16h bva, bvb;
    {
      const _Float16* va2 = vbase + (size_t)m * L_ + j0 + hi * 16;
      v8h t0 = *(const v8h*)va2, t1 = *(const v8h*)(va2 + 8);
      const _Float16* vb2 = vbase + (size_t)(m + 16) * L_ + j0 + hi * 16;
      v8h u0 = *(const v8h*)vb2, u1 = *(const v8h*)(vb2 + 8);
#pragma unroll
      for (int t = 0; t < 8; ++t) {
        bva[t] = t0[t]; bva[8 + t] = t1[t];
        bvb[t] = u0[t]; bvb[8 + t] = u1[t];
      }
    }
    oa = __builtin_amdgcn_wmma_f32_16x16x32_f16(false, ap, false, bva,
                                                (short)0, oa, false, false);
    ob = __builtin_amdgcn_wmma_f32_16x16x32_f16(false, ap, false, bvb,
                                                (short)0, ob, false, false);
  }

  // ---- epilogue: normalize and scatter to AO[bh*32 + d][i] (f32, c-major)
  float* aob = AO + (size_t)bh * 32 * L_;
#pragma unroll
  for (int r = 0; r < 8; ++r) {
    const float ri = 1.0f / rsum[r];
    const int irow = i0 + r + 8 * hi;
    aob[(size_t)m * L_ + irow]        = oa[r] * ri;
    aob[(size_t)(m + 16) * L_ + irow] = ob[r] * ri;
  }
}

// ============================================================================
// Kernel 5: output projection  y[b][o][l] = sum_c w_out[o][c]*AO[b][c][l] + b_out[o]
// One wave per 16x16 tile, v_wmma_f32_16x16x4_f32 over K=128.
// waves = B_ * 16 * 256 = 8192 -> 1024 blocks.
// ============================================================================
__global__ __launch_bounds__(256) void out_gemm_kernel(const float* __restrict__ AO,
                                                       const float* __restrict__ w,
                                                       const float* __restrict__ bias,
                                                       float* __restrict__ y) {
  const int wg   = blockIdx.x * 8 + (threadIdx.x >> 5);
  const int lane = threadIdx.x & 31;
  const int m    = lane & 15;
  const int hi   = lane >> 4;
  const int lt   = wg & 255;
  const int ot   = (wg >> 8) & 15;
  const int b    = wg >> 12;

  const float* arow = w + (ot * 16 + m) * HID_ + hi * 2;
  const float* bcol = AO + (size_t)b * HID_ * L_ + (lt * 16 + m);

  v8f acc = {};
#pragma unroll 4
  for (int k0 = 0; k0 < HID_; k0 += 4) {
    v2f a = *(const v2f*)(arow + k0);
    v2f bb;
    bb[0] = bcol[(size_t)(k0 + hi * 2) * L_];
    bb[1] = bcol[(size_t)(k0 + hi * 2 + 1) * L_];
    acc = __builtin_amdgcn_wmma_f32_16x16x4_f32(false, a, false, bb,
                                                (short)0, acc, false, false);
  }
#pragma unroll
  for (int r = 0; r < 8; ++r) {
    const int o = ot * 16 + r + 8 * hi;
    y[((size_t)b * C_ + o) * L_ + lt * 16 + m] = acc[r] + bias[o];
  }
}

// ============================================================================
extern "C" void kernel_launch(void* const* d_in, const int* in_sizes, int n_in,
                              void* d_out, int out_size, void* d_ws, size_t ws_size,
                              hipStream_t stream) {
  const float* x     = (const float*)d_in[0];   // (2,256,4096)
  const float* w_qkv = (const float*)d_in[1];   // (384,256)
  const float* w_out = (const float*)d_in[2];   // (256,128)
  const float* b_out = (const float*)d_in[3];   // (256,)
  float* y = (float*)d_out;                     // (2,256,4096)

  char* ws = (char*)d_ws;
  float*    G    = (float*)(ws);                   // 2*384*4096*4  = 12,582,912 B
  float*    invn = (float*)(ws + 12582912);        // 2*256*4      = 2,048 B
  _Float16* qt   = (_Float16*)(ws + 12584960);     // 2*4*4096*32*2 = 2,097,152 B
  _Float16* kt   = (_Float16*)(ws + 14682112);     // 2,097,152 B
  _Float16* vv   = (_Float16*)(ws + 16779264);     // 2,097,152 B
  float*    AO   = (float*)(ws + 18876416);        // 2*128*4096*4 = 4,194,304 B
  (void)in_sizes; (void)n_in; (void)out_size; (void)ws_size;

  qkv_gemm_kernel<<<1536, 256, 0, stream>>>(x, w_qkv, G);
  norm_kernel    <<<512,  256, 0, stream>>>(G, invn);
  pack_kernel    <<<4096, 256, 0, stream>>>(G, invn, qt, kt, vv);
  flash_kernel   <<<256,  256, 0, stream>>>(qt, kt, vv, AO);
  out_gemm_kernel<<<1024, 256, 0, stream>>>(AO, w_out, b_out, y);
}
